// SimplifiedMamba_36232344109110
// MI455X (gfx1250) — matrix operand order repaired and checked by reading, not compiled, not run on hardware
//
#include <hip/hip_runtime.h>
#include <math.h>

// ---- problem constants (match reference) ----
#define D_MODEL 1024
#define D_STATE 16
#define D_CONV  4
#define D_INNER 2048
#define BATCH   2
#define SEQ     2048
#define ML      (BATCH * SEQ)   // 4096 "token rows"

typedef __bf16 bf16;
typedef __attribute__((ext_vector_type(16))) __bf16 bf16x16;
typedef __attribute__((ext_vector_type(8)))  __bf16 bf16x8;
typedef __attribute__((ext_vector_type(8)))  float  f32x8;
typedef int v4i __attribute__((__vector_size__(4 * sizeof(int))));

__device__ __forceinline__ float silu_f(float x) { return x / (1.0f + expf(-x)); }
__device__ __forceinline__ float softplus_f(float x) {
    return (x > 20.0f) ? x : log1pf(expf(x));
}

// ---------------------------------------------------------------------------
// Async global->LDS staging (gfx1250 GLOBAL_LOAD_ASYNC_TO_LDS_B128, ASYNCcnt).
// Toolchain-probed signature: (v4i as1*, v4i as3*, int offset, int cpol).
// Falls back to sync VGPR staging if the builtin is not declared.
// ---------------------------------------------------------------------------
#if defined(__has_builtin)
#  if __has_builtin(__builtin_amdgcn_global_load_async_to_lds_b128)
#    define USE_ASYNC_LDS 1
#  endif
#endif

__device__ __forceinline__ void stage16(bf16* ldst, const bf16* gsrc) {
#ifdef USE_ASYNC_LDS
    __builtin_amdgcn_global_load_async_to_lds_b128(
        (__attribute__((address_space(1))) v4i*)(uintptr_t)gsrc,
        (__attribute__((address_space(3))) v4i*)(uint32_t)(uintptr_t)ldst,
        0, 0);
#else
    *(uint4*)ldst = *(const uint4*)gsrc;
#endif
}

__device__ __forceinline__ void stage_wait() {
#ifdef USE_ASYNC_LDS
#  if __has_builtin(__builtin_amdgcn_s_wait_asynccnt)
    __builtin_amdgcn_s_wait_asynccnt(0);
#  else
    asm volatile("s_wait_asynccnt 0x0" ::: "memory");
#  endif
#endif
}

// ============================================================================
// Tiled bf16 WMMA GEMM:  C[M,N] = A[M,K](bf16) @ W[N,K](bf16)^T  (+ epilogue)
//   MODE 0: plain fp32 store
//   MODE 1: softplus(acc + bias[col])
// Block = 256 threads = 8 waves (2 in M x 4 in N); block tile 64x128, K=64.
// Per wave: 32x32 output = 2x2 f32x8 accumulators; 8 WMMAs per K-chunk.
// LDS rows padded to 72 bf16 (144 B) so every 16-B fragment slice is aligned.
// ============================================================================
template <int MODE>
__global__ __launch_bounds__(256) void gemm_bf16_wmma(
    const bf16* __restrict__ A, const bf16* __restrict__ W,
    const float* __restrict__ bias, float* __restrict__ C,
    int M, int N, int K, int lda, int ldc)
{
    __shared__ bf16 smA[64 * 72];
    __shared__ bf16 smB[128 * 72];

    const int tid    = threadIdx.x;
    const int wave   = tid >> 5;
    const int lane   = tid & 31;
    const int lane15 = lane & 15;
    const int waveM  = wave & 1;   // 0..1  (32-row strips)
    const int waveN  = wave >> 1;  // 0..3  (32-col strips)
    const int m0 = blockIdx.x * 64;
    const int n0 = blockIdx.y * 128;

    f32x8 acc[2][2] = {};

    for (int k0 = 0; k0 < K; k0 += 64) {
        // ---- stage A tile 64x64 bf16: 512 16-B chunks, 2 per thread
        #pragma unroll
        for (int it = 0; it < 2; ++it) {
            int c = tid + it * 256;
            int row = c >> 3, kc = (c & 7) << 3;
            stage16(&smA[row * 72 + kc], A + (size_t)(m0 + row) * lda + k0 + kc);
        }
        // ---- stage B tile 128x64 bf16: 1024 16-B chunks, 4 per thread
        #pragma unroll
        for (int it = 0; it < 4; ++it) {
            int c = tid + it * 256;
            int row = c >> 3, kc = (c & 7) << 3;
            stage16(&smB[row * 72 + kc], W + (size_t)(n0 + row) * K + k0 + kc);
        }
        // prefetch next K-chunk into cache (global_prefetch_b8)
        if (k0 + 64 < K) {
            __builtin_prefetch(A + (size_t)(m0 + (tid >> 2)) * lda + k0 + 64, 0, 3);
            __builtin_prefetch(W + (size_t)(n0 + (tid >> 1)) * K + k0 + 64, 0, 3);
        }
        stage_wait();
        __syncthreads();

        #pragma unroll
        for (int kk = 0; kk < 64; kk += 32) {
            // A fragments: 16x32 bf16; lane M = lane&15,
            // element e -> K = kk + (lane>=16?8:0) + (e>=8?16:0) + (e&7)
            bf16x16 af[2];
            #pragma unroll
            for (int i = 0; i < 2; ++i) {
                const bf16* rp = &smA[(waveM * 32 + i * 16 + lane15) * 72];
                const int klo = (lane & 16) ? 8 : 0;
                const bf16x8 lo = *(const bf16x8*)(rp + kk + klo);
                const bf16x8 hi = *(const bf16x8*)(rp + kk + klo + 16);
                #pragma unroll
                for (int e = 0; e < 8; ++e) { af[i][e] = lo[e]; af[i][8 + e] = hi[e]; }
            }
            // B fragments: 32x16 bf16; lane N = lane&15,
            // element e -> K = kk + (lane>=16?8:0) + (e>=8?16:0) + (e&7)
            bf16x16 bfr[2];
            #pragma unroll
            for (int j = 0; j < 2; ++j) {
                const bf16* rp = &smB[(waveN * 32 + j * 16 + lane15) * 72];
                const int kb = (lane & 16) ? 8 : 0;
                const bf16x8 lo = *(const bf16x8*)(rp + kk + kb);
                const bf16x8 hi = *(const bf16x8*)(rp + kk + kb + 16);
                #pragma unroll
                for (int e = 0; e < 8; ++e) { bfr[j][e] = lo[e]; bfr[j][8 + e] = hi[e]; }
            }
            #pragma unroll
            for (int i = 0; i < 2; ++i)
                #pragma unroll
                for (int j = 0; j < 2; ++j)
                    acc[i][j] = __builtin_amdgcn_wmma_f32_16x16x32_bf16(
                        false, af[i], false, bfr[j], (short)0, acc[i][j], false, false);
        }
        __syncthreads();
    }

    // ---- epilogue: D layout: vgpr r -> row r + (lane>=16?8:0), col = lane&15
    const int rsel = (lane < 16) ? 0 : 8;
    #pragma unroll
    for (int i = 0; i < 2; ++i) {
        #pragma unroll
        for (int j = 0; j < 2; ++j) {
            const int col  = n0 + waveN * 32 + j * 16 + lane15;
            const int row0 = m0 + waveM * 32 + i * 16 + rsel;
            const float bv = (MODE == 1) ? bias[col] : 0.0f;
            #pragma unroll
            for (int r = 0; r < 8; ++r) {
                float v = acc[i][j][r];
                if (MODE == 1) v = softplus_f(v + bv);
                C[(size_t)(row0 + r) * ldc + col] = v;
            }
        }
    }
}

// ============================================================================
// fp32 -> bf16 bulk convert (weights once, activations once per pass).
// ============================================================================
__global__ __launch_bounds__(256) void cvt_f32_bf16_kernel(
    const float* __restrict__ src, bf16* __restrict__ dst)
{
    int i = blockIdx.x * blockDim.x + threadIdx.x;
    float4 v = ((const float4*)src)[i];
    bf16* p = dst + (size_t)i * 4;
    p[0] = (bf16)v.x; p[1] = (bf16)v.y; p[2] = (bf16)v.z; p[3] = (bf16)v.w;
}

// ============================================================================
// Causal depthwise conv1d (width 4, left pad 3) + bias + SiLU.
// Emits fp32 x_act (scan/bc consumers) and bf16 x_act (dt_proj GEMM A).
// ============================================================================
__global__ __launch_bounds__(256) void conv_silu_kernel(
    const float* __restrict__ xz, const float* __restrict__ cw,
    const float* __restrict__ cb, float* __restrict__ xact,
    bf16* __restrict__ xactb)
{
    int idx = blockIdx.x * blockDim.x + threadIdx.x;   // ML * D_INNER
    int d = idx & (D_INNER - 1);
    int t = idx >> 11;                                 // token row (b*SEQ + l)
    int l = t & (SEQ - 1);
    float s = cb[d];
    #pragma unroll
    for (int j = 0; j < D_CONV; ++j) {
        int li = l - (D_CONV - 1) + j;
        if (li >= 0)
            s += cw[d * D_CONV + j] *
                 xz[(size_t)(t - (D_CONV - 1) + j) * (2 * D_INNER) + d];
    }
    float v = silu_f(s);
    xact[(size_t)t * D_INNER + d]  = v;
    xactb[(size_t)t * D_INNER + d] = (bf16)v;
}

// ============================================================================
// bc = x_act @ x_proj_w.T  (N = 32 only -> one wave per token row, lane = n).
// ============================================================================
__global__ void bc_kernel(const float* __restrict__ xact,
                          const float* __restrict__ xw,   // [32][D_INNER]
                          float* __restrict__ bc)
{
    int m = blockIdx.x;
    int n = threadIdx.x;            // 0..31
    const float* xr = xact + (size_t)m * D_INNER;
    const float* wr = xw + (size_t)n * D_INNER;
    float acc = 0.0f;
    #pragma unroll 4
    for (int k = 0; k < D_INNER; k += 4)
        acc += xr[k] * wr[k] + xr[k + 1] * wr[k + 1] +
               xr[k + 2] * wr[k + 2] + xr[k + 3] * wr[k + 3];
    bc[m * 32 + n] = acc;
}

// ============================================================================
// Selective scan: lane = one (b, d, n) state; 16 lanes form one channel.
// Sequential over L; y = sum_n h*C via shfl_xor 1/2/4/8 (wave32 half-groups).
// ============================================================================
__global__ __launch_bounds__(256) void scan_kernel(
    const float* __restrict__ delta, const float* __restrict__ xact,
    const float* __restrict__ bc, const float* __restrict__ A_log,
    const float* __restrict__ Dp, float* __restrict__ yssm)
{
    int gtid = blockIdx.x * blockDim.x + threadIdx.x;  // BATCH*D_INNER*D_STATE
    int n = gtid & (D_STATE - 1);
    int c = gtid >> 4;
    int d = c & (D_INNER - 1);
    int b = c >> 11;
    const float Ad = -expf(A_log[d * D_STATE + n]);
    const float dparam = Dp[d];
    float h = 0.0f;
    for (int l = 0; l < SEQ; ++l) {
        size_t row = (size_t)b * SEQ + l;
        float dt = delta[row * D_INNER + d];
        float u  = xact[row * D_INNER + d];
        float Bv = bc[row * 32 + n];
        float Cv = bc[row * 32 + D_STATE + n];
        h = expf(dt * Ad) * h + (dt * u) * Bv;
        float y = h * Cv;
        y += __shfl_xor(y, 1);
        y += __shfl_xor(y, 2);
        y += __shfl_xor(y, 4);
        y += __shfl_xor(y, 8);
        if (n == 0) yssm[row * D_INNER + d] = y + u * dparam;
    }
}

// ============================================================================
// Gate: ygate = bf16(y_ssm * silu(z)); z lives in xz[:, D_INNER:2*D_INNER].
// ============================================================================
__global__ __launch_bounds__(256) void gate_kernel(
    const float* __restrict__ yssm, const float* __restrict__ xz,
    bf16* __restrict__ ygate)
{
    int idx = blockIdx.x * blockDim.x + threadIdx.x;   // ML * D_INNER
    int d = idx & (D_INNER - 1);
    int t = idx >> 11;
    float z = xz[(size_t)t * (2 * D_INNER) + D_INNER + d];
    ygate[idx] = (bf16)(yssm[idx] * silu_f(z));
}

// ============================================================================
extern "C" void kernel_launch(void* const* d_in, const int* in_sizes, int n_in,
                              void* d_out, int out_size, void* d_ws, size_t ws_size,
                              hipStream_t stream)
{
    const float* x    = (const float*)d_in[0];
    const float* w_in = (const float*)d_in[1];
    const float* cw   = (const float*)d_in[2];
    const float* cb   = (const float*)d_in[3];
    const float* xw   = (const float*)d_in[4];
    const float* dtw  = (const float*)d_in[5];
    const float* dtb  = (const float*)d_in[6];
    const float* alog = (const float*)d_in[7];
    const float* dpar = (const float*)d_in[8];
    const float* wout = (const float*)d_in[9];
    float* out = (float*)d_out;

    // ---- workspace layout (fp32 region, then bf16 region) ----
    float* ws    = (float*)d_ws;
    float* xz    = ws;                                   // [ML][2*D_INNER]
    float* xact  = xz    + (size_t)ML * 2 * D_INNER;     // [ML][D_INNER]
    float* delta = xact  + (size_t)ML * D_INNER;         // [ML][D_INNER]
    float* bcbuf = delta + (size_t)ML * D_INNER;         // [ML][32]
    float* yssm  = bcbuf + (size_t)ML * 2 * D_STATE;     // [ML][D_INNER]
    bf16* bws      = (bf16*)(yssm + (size_t)ML * D_INNER);
    bf16* x_bf     = bws;                                // [ML][D_MODEL]
    bf16* w_in_bf  = x_bf    + (size_t)ML * D_MODEL;     // [2*D_INNER][D_MODEL]
    bf16* dtw_bf   = w_in_bf + (size_t)2 * D_INNER * D_MODEL;  // [D_INNER][D_INNER]
    bf16* wout_bf  = dtw_bf  + (size_t)D_INNER * D_INNER;      // [D_MODEL][D_INNER]
    bf16* xact_bf  = wout_bf + (size_t)D_MODEL * D_INNER;      // [ML][D_INNER]
    bf16* ygate_bf = xact_bf + (size_t)ML * D_INNER;           // [ML][D_INNER]

    // 0) one-shot fp32 -> bf16 converts (weights + x)
    cvt_f32_bf16_kernel<<<(ML * D_MODEL) / 1024, 256, 0, stream>>>(x, x_bf);
    cvt_f32_bf16_kernel<<<(2 * D_INNER * D_MODEL) / 1024, 256, 0, stream>>>(w_in, w_in_bf);
    cvt_f32_bf16_kernel<<<(D_INNER * D_INNER) / 1024, 256, 0, stream>>>(dtw, dtw_bf);
    cvt_f32_bf16_kernel<<<(D_MODEL * D_INNER) / 1024, 256, 0, stream>>>(wout, wout_bf);

    // 1) xz = x @ in_proj_w.T             (M=4096, N=4096, K=1024)
    gemm_bf16_wmma<0><<<dim3(ML / 64, (2 * D_INNER) / 128), 256, 0, stream>>>(
        x_bf, w_in_bf, nullptr, xz, ML, 2 * D_INNER, D_MODEL, D_MODEL, 2 * D_INNER);

    // 2) x_act = silu(conv1d(x_ssm) + b)  (fp32 + bf16 copies)
    conv_silu_kernel<<<(ML * D_INNER) / 256, 256, 0, stream>>>(xz, cw, cb, xact, xact_bf);

    // 3) delta = softplus(x_act @ dt_proj_w.T + dt_proj_b)  (4096x2048x2048)
    gemm_bf16_wmma<1><<<dim3(ML / 64, D_INNER / 128), 256, 0, stream>>>(
        xact_bf, dtw_bf, dtb, delta, ML, D_INNER, D_INNER, D_INNER, D_INNER);

    // 4) bc = x_act @ x_proj_w.T          (N=32 GEMV-ish, fp32)
    bc_kernel<<<ML, 32, 0, stream>>>(xact, xw, bcbuf);

    // 5) selective scan -> y_ssm (fp32 state)
    scan_kernel<<<(BATCH * D_INNER * D_STATE) / 256, 256, 0, stream>>>(
        delta, xact, bcbuf, alog, dpar, yssm);

    // 6) gate: ygate_bf = bf16(y_ssm * silu(z))
    gate_kernel<<<(ML * D_INNER) / 256, 256, 0, stream>>>(yssm, xz, ygate_bf);

    // 7) out = ygate @ out_proj_w.T       (M=4096, N=1024, K=2048)
    gemm_bf16_wmma<0><<<dim3(ML / 64, D_MODEL / 128), 256, 0, stream>>>(
        ygate_bf, wout_bf, nullptr, out, ML, D_MODEL, D_INNER, D_INNER, D_MODEL);
}
